// NTM_15839839387874
// MI455X (gfx1250) — matrix operand order repaired and checked by reading, not compiled, not run on hardware
//
#include <hip/hip_runtime.h>
#include <math.h>

#define B_   128
#define T_   64
#define D_   64
#define SU_  256
#define O_   64
#define H_   4
#define M_   64
#define N_   1024
#define NS_  3
#define P_   (3*M_ + NS_ + 3)   // 198
#define HP_  (H_*P_)            // 792
#define NT   256                // threads / block (8 waves of 32)
#define NW   8

typedef __attribute__((ext_vector_type(16))) _Float16 v16h;
typedef __attribute__((ext_vector_type(8)))  float    v8f;
typedef __attribute__((vector_size(16)))     int      v4i_;

#if defined(__has_builtin)
#if __has_builtin(__builtin_amdgcn_global_load_async_to_lds_b128) && \
    __has_builtin(__builtin_amdgcn_s_wait_asynccnt)
#define HAVE_ASYNC_LDS 1
#endif
#endif

// LDS layout (floats)
#define LDS_FLOATS (N_*M_ + H_*N_ + H_*N_ + N_ + 2048 + SU_ + H_*M_ + HP_ + \
                    H_*M_ + H_*M_ + D_ + NT + 64)   // = 79000 floats = 316 KB

__device__ __forceinline__ float sigmoidf_(float v)  { return 1.f / (1.f + __expf(-v)); }
__device__ __forceinline__ float softplusf_(float v) { return (v > 20.f) ? v : log1pf(__expf(v)); }

// ---- WMMA fragment loaders (16-bit, contiguous k order shared by A and B; not the ISA table order) ----
// A (16x32, f16): lane L holds row = L%16; element e=2v+j maps to
//   K = e + 16*(L/16)
// Only rows < H_ are real; rows 4..15 are zero-padded BRANCHLESSLY:
// read row&3 (always a valid address -> unconditional, mergeable DS load)
// and multiply by a per-lane 0/1 mask.
__device__ __forceinline__ v16h load_a_4row(const float* src, int ld,
                                            int kb, int lane) {
  int row  = lane & 15, half = lane >> 4;
  float msk = (row < H_) ? 1.f : 0.f;
  const float* rp = src + (row & (H_ - 1)) * ld + kb;
  v16h a;
#pragma unroll
  for (int e = 0; e < 16; ++e) {
    int v = e >> 1, j = e & 1;
    int kk = 2 * v + j + 16 * half;
    a[e] = (_Float16)(rp[kk] * msk);
  }
  return a;
}
// B (32x16, f16): lane L holds col = L%16; element e maps to K = e + 16*(L/16)
// B[k][col] = src[(kb+k)*ld + cb + col]   (K-major source, e.g. mem rows = K)
__device__ __forceinline__ v16h load_b_rowmajor(const float* src, int ld,
                                                int kb, int cb, int lane) {
  int col = lane & 15, half = lane >> 4;
  v16h b;
#pragma unroll
  for (int e = 0; e < 16; ++e) {
    int kk = e + 16 * half;
    b[e] = (_Float16)src[(kb + kk) * ld + cb + col];
  }
  return b;
}
// B[k][col] = src[(cb+col)*ld + kb + k]   (transposed source, e.g. mem^T)
__device__ __forceinline__ v16h load_b_colmajor(const float* src, int ld,
                                                int kb, int cb, int lane) {
  int col = lane & 15, half = lane >> 4;
  v16h b;
#pragma unroll
  for (int e = 0; e < 16; ++e) {
    int kk = e + 16 * half;
    b[e] = (_Float16)src[(cb + col) * ld + kb + kk];
  }
  return b;
}

__global__ __launch_bounds__(NT, 1)
void ntm_scan_kernel(const float* __restrict__ x,    const float* __restrict__ ctrl0,
                     const float* __restrict__ wt0,  const float* __restrict__ mem0,
                     const float* __restrict__ Wxr,  const float* __restrict__ Whh,
                     const float* __restrict__ bh,   const float* __restrict__ Wo,
                     const float* __restrict__ bo,   const float* __restrict__ Wp,
                     const float* __restrict__ bp,   float* __restrict__ out) {
  extern __shared__ float lds[];
  float* mem_s  = lds;                    // N*M   = 65536
  float* w_s    = mem_s  + N_ * M_;       // H*N   = 4096
  float* w2_s   = w_s    + H_ * N_;       // H*N   = 4096 (sim/wc/wg/wpow scratch)
  float* norm_s = w2_s   + H_ * N_;       // N     = 1024 (1/(||mem_n||+eps))
  float* rpart  = norm_s + N_;            // 2*4*256 = 2048 (read partials)
  float* h_s    = rpart  + 2048;          // SU    = 256
  float* r_s    = h_s    + SU_;           // H*M   = 256
  float* p_s    = r_s    + H_ * M_;       // H*P   = 792
  float* e_s    = p_s    + HP_;           // H*M   = 256 (erase, sigmoided)
  float* kn_s   = e_s    + H_ * M_;       // H*M   = 256 (normalized keys)
  float* x_s    = kn_s   + H_ * M_;       // D     = 64
  float* red_s  = x_s    + D_;            // NT    = 256 (reductions)
  float* prm_s  = red_s  + NT;            // 64    (small per-head params)

  float* hp_knorm = prm_s;        // 4
  float* hp_beta  = prm_s + 4;    // 4
  float* hp_g     = prm_s + 8;    // 4
  float* hp_gamma = prm_s + 12;   // 4
  float* hp_sv    = prm_s + 16;   // 12 (shift weights)
  float* hp_hmax  = prm_s + 28;   // 4
  float* hp_hsum  = prm_s + 32;   // 4

  const int bb_  = blockIdx.x;       // batch element
  const int tid  = threadIdx.x;
  const int lane = tid & 31;
  const int wid  = tid >> 5;

  // Pull shared weights toward the caches (global_prefetch_b8).
  if (tid == 0) {
    __builtin_prefetch(Wxr, 0, 1);
    __builtin_prefetch(Whh, 0, 1);
    __builtin_prefetch(Wp,  0, 1);
    __builtin_prefetch(Wo,  0, 1);
  }

  // ---- initial state -> LDS ----
  const float* memg = mem0 + (size_t)bb_ * (N_ * M_);
#ifdef HAVE_ASYNC_LDS
  for (int i = tid; i < (N_ * M_) / 4; i += NT) {
    __builtin_amdgcn_global_load_async_to_lds_b128(
        (v4i_*)(memg + 4 * i),
        (v4i_*)(mem_s + 4 * i), 0, 0);
  }
  __builtin_amdgcn_s_wait_asynccnt(0);
#else
  for (int i = tid; i < (N_ * M_) / 4; i += NT)
    ((float4*)mem_s)[i] = ((const float4*)memg)[i];
#endif
  for (int i = tid; i < (H_ * N_) / 4; i += NT)
    ((float4*)w_s)[i] = ((const float4*)(wt0 + (size_t)bb_ * H_ * N_))[i];
  if (tid < SU_ / 4)
    ((float4*)h_s)[tid] = ((const float4*)(ctrl0 + (size_t)bb_ * SU_))[tid];
  __syncthreads();

  for (int t = 0; t < T_; ++t) {
    // ---- x_t and mem row norms ----
    if (tid < D_) x_s[tid] = x[((size_t)bb_ * T_ + t) * D_ + tid];
    for (int n = tid; n < N_; n += NT) {
      const float* row = mem_s + n * M_;
      float s = 0.f;
#pragma unroll 8
      for (int m = 0; m < M_; ++m) { float v = row[m]; s += v * v; }
      norm_s[n] = 1.f / (sqrtf(s) + 1e-8f);
    }
    __syncthreads();

    // ---- Phase A: r = w (H x N) * mem (N x M) via WMMA, K=1024 split into 2 halves ----
    {
      const int c  = wid & 3;   // 16-col tile of M
      const int kh = wid >> 2;  // K half
      v8f acc = {};
      for (int kc = 0; kc < 16; ++kc) {
        int kb = kh * 512 + kc * 32;
        v16h af = load_a_4row(w_s, N_, kb, lane);
        v16h bf = load_b_rowmajor(mem_s, M_, kb, c * 16, lane);
        acc = __builtin_amdgcn_wmma_f32_16x16x32_f16(false, af, false, bf,
                                                     (short)0, acc, false, false);
      }
      float* dst = rpart + (kh * 4 + c) * 256;
      int col = lane & 15, half = lane >> 4;
#pragma unroll
      for (int v = 0; v < 8; ++v) dst[(v + 8 * half) * 16 + col] = acc[v];
    }
    __syncthreads();
    {  // combine the two K-half partials (only heads 0..3 are real rows)
      int hh = tid >> 6, m = tid & 63;
      int c = m >> 4, cc = m & 15;
      r_s[hh * M_ + m] = rpart[c * 256 + hh * 16 + cc] +
                         rpart[(4 + c) * 256 + hh * 16 + cc];
    }
    __syncthreads();

    // ---- Phase B: controller (GEMVs -> VALU) ----
    {
      const int s = tid;  // one output column per thread
      float pre = bh[s];
#pragma unroll 4
      for (int i = 0; i < D_; ++i)      pre += x_s[i] * Wxr[i * SU_ + s];
#pragma unroll 4
      for (int i = 0; i < H_ * M_; ++i) pre += r_s[i] * Wxr[(D_ + i) * SU_ + s];
#pragma unroll 4
      for (int i = 0; i < SU_; ++i)     pre += h_s[i] * Whh[i * SU_ + s];
      float hn = tanhf(pre);
      __syncthreads();
      h_s[s] = hn;
      __syncthreads();
    }
    if (tid < O_) {
      float acc = bo[tid];
      for (int i = 0; i < SU_; ++i) acc += h_s[i] * Wo[i * O_ + tid];
      out[((size_t)bb_ * T_ + t) * O_ + tid] = acc;
    }
    for (int pc = tid; pc < HP_; pc += NT) {
      float acc = bp[pc];
#pragma unroll 4
      for (int i = 0; i < SU_; ++i) acc += h_s[i] * Wp[i * HP_ + pc];
      p_s[pc] = acc;
    }
    __syncthreads();

    // ---- Phase C: per-head params ----
    if (tid < H_) {
      const float* ph = p_s + tid * P_;
      float kn2 = 0.f;
      for (int m = 0; m < M_; ++m) kn2 += ph[m] * ph[m];
      hp_knorm[tid] = 1.f / (sqrtf(kn2) + 1e-8f);
      hp_beta[tid]  = softplusf_(ph[M_]);
      hp_g[tid]     = sigmoidf_(ph[M_ + 1]);
      float s0 = ph[M_ + 2], s1 = ph[M_ + 3], s2 = ph[M_ + 4];
      float mx = fmaxf(s0, fmaxf(s1, s2));
      float e0 = __expf(s0 - mx), e1 = __expf(s1 - mx), e2 = __expf(s2 - mx);
      float inv = 1.f / (e0 + e1 + e2);
      hp_sv[tid * 3 + 0] = e0 * inv;
      hp_sv[tid * 3 + 1] = e1 * inv;
      hp_sv[tid * 3 + 2] = e2 * inv;
      hp_gamma[tid] = 1.f + softplusf_(ph[M_ + 2 + NS_]);
    }
    __syncthreads();
    {
      int hh = tid >> 6, m = tid & 63;
      const float* ph = p_s + hh * P_;
      kn_s[hh * M_ + m] = ph[m] * hp_knorm[hh];
      e_s[hh * M_ + m]  = sigmoidf_(ph[M_ + 3 + NS_ + m]);
    }
    __syncthreads();

    // ---- Phase D: sim = kn (HxM) * mem^T (MxN) via WMMA; fused *beta*invnorm ----
    {
      v16h a0 = load_a_4row(kn_s, M_, 0,  lane);
      v16h a1 = load_a_4row(kn_s, M_, 32, lane);
      int col = lane & 15, half = lane >> 4;
      for (int tile = wid; tile < N_ / 16; tile += NW) {
        v8f acc = {};
        v16h b0 = load_b_colmajor(mem_s, M_, 0,  tile * 16, lane);
        acc = __builtin_amdgcn_wmma_f32_16x16x32_f16(false, a0, false, b0,
                                                     (short)0, acc, false, false);
        v16h b1 = load_b_colmajor(mem_s, M_, 32, tile * 16, lane);
        acc = __builtin_amdgcn_wmma_f32_16x16x32_f16(false, a1, false, b1,
                                                     (short)0, acc, false, false);
        // rows 0..3 live in VGPRs 0..3 of the lanes with half==0
        if (half == 0) {
          int n = tile * 16 + col;
          float iv = norm_s[n];
#pragma unroll
          for (int v = 0; v < H_; ++v)
            w2_s[v * N_ + n] = hp_beta[v] * acc[v] * iv;
        }
      }
    }
    __syncthreads();

    // ---- softmax over N per head -> wc ; then wg = g*wc + (1-g)*w ----
    {
      int hh = tid >> 6, j = tid & 63;  // 64 threads per head
      float mx = -1e30f;
      for (int n = j; n < N_; n += 64) mx = fmaxf(mx, w2_s[hh * N_ + n]);
      red_s[tid] = mx;
      __syncthreads();
      if (tid < H_) {
        float m2 = -1e30f;
        for (int q = 0; q < 64; ++q) m2 = fmaxf(m2, red_s[tid * 64 + q]);
        hp_hmax[tid] = m2;
      }
      __syncthreads();
      float hm = hp_hmax[hh];
      float sum = 0.f;
      for (int n = j; n < N_; n += 64) {
        float v = __expf(w2_s[hh * N_ + n] - hm);
        w2_s[hh * N_ + n] = v;
        sum += v;
      }
      red_s[tid] = sum;
      __syncthreads();
      if (tid < H_) {
        float s2 = 0.f;
        for (int q = 0; q < 64; ++q) s2 += red_s[tid * 64 + q];
        hp_hsum[tid] = 1.f / s2;
      }
      __syncthreads();
      float g = hp_g[hh], invs = hp_hsum[hh];
      for (int n = j; n < N_; n += 64) {
        float wc = w2_s[hh * N_ + n] * invs;
        w2_s[hh * N_ + n] = g * wc + (1.f - g) * w_s[hh * N_ + n];
      }
    }
    __syncthreads();

    // ---- circular shift + sharpen (register-staged to do it in place) ----
    {
      int hh = tid >> 6, j = tid & 63;
      float s0 = hp_sv[hh * 3 + 0], s1 = hp_sv[hh * 3 + 1], s2 = hp_sv[hh * 3 + 2];
      float gam = hp_gamma[hh];
      float vals[16];
#pragma unroll
      for (int q = 0; q < 16; ++q) {
        int n  = j + q * 64;
        int np = (n + 1) & (N_ - 1), nm = (n - 1) & (N_ - 1);
        vals[q] = s0 * w2_s[hh * N_ + np] + s1 * w2_s[hh * N_ + n] +
                  s2 * w2_s[hh * N_ + nm];
      }
      __syncthreads();
#pragma unroll
      for (int q = 0; q < 16; ++q) {
        int n = j + q * 64;
        float v = fmaxf(vals[q], 1e-12f);
        w2_s[hh * N_ + n] = __expf(gam * __logf(v));
      }
    }
    __syncthreads();
    {  // normalize -> new w
      int hh = tid >> 6, j = tid & 63;
      float sum = 0.f;
      for (int n = j; n < N_; n += 64) sum += w2_s[hh * N_ + n];
      red_s[tid] = sum;
      __syncthreads();
      if (tid < H_) {
        float s2 = 0.f;
        for (int q = 0; q < 64; ++q) s2 += red_s[tid * 64 + q];
        hp_hsum[tid] = 1.f / s2;
      }
      __syncthreads();
      float invs = hp_hsum[hh];
      for (int n = j; n < N_; n += 64) w_s[hh * N_ + n] = w2_s[hh * N_ + n] * invs;
    }
    __syncthreads();

    // ---- Phase E: mem = mem * prod_h(1 - w_h e_h) + sum_h w_h a_h ----
    {
      const int aoff = M_ + 3 + NS_ + M_;  // 134
      for (int i = tid; i < N_ * M_; i += NT) {
        int n = i >> 6, m = i & 63;
        float w0 = w_s[0 * N_ + n], w1 = w_s[1 * N_ + n];
        float w2v = w_s[2 * N_ + n], w3 = w_s[3 * N_ + n];
        float keep = (1.f - w0 * e_s[0 * M_ + m]) * (1.f - w1 * e_s[1 * M_ + m]) *
                     (1.f - w2v * e_s[2 * M_ + m]) * (1.f - w3 * e_s[3 * M_ + m]);
        float add = w0 * p_s[0 * P_ + aoff + m] + w1 * p_s[1 * P_ + aoff + m] +
                    w2v * p_s[2 * P_ + aoff + m] + w3 * p_s[3 * P_ + aoff + m];
        mem_s[i] = mem_s[i] * keep + add;
      }
    }
    __syncthreads();
  }  // t

  // ---- final state writeback: out = [output | h | w | mem] ----
  float* out_h = out + (size_t)B_ * T_ * O_;
  float* out_w = out_h + (size_t)B_ * SU_;
  float* out_m = out_w + (size_t)B_ * H_ * N_;
  if (tid < SU_) out_h[(size_t)bb_ * SU_ + tid] = h_s[tid];
  for (int i = tid; i < H_ * N_; i += NT) out_w[(size_t)bb_ * H_ * N_ + i] = w_s[i];
  for (int i = tid; i < N_ * M_; i += NT) out_m[(size_t)bb_ * N_ * M_ + i] = mem_s[i];
}

extern "C" void kernel_launch(void* const* d_in, const int* in_sizes, int n_in,
                              void* d_out, int out_size, void* d_ws, size_t ws_size,
                              hipStream_t stream) {
  (void)in_sizes; (void)n_in; (void)out_size; (void)d_ws; (void)ws_size;
  const float* x     = (const float*)d_in[0];
  const float* ctrl0 = (const float*)d_in[1];
  const float* wt0   = (const float*)d_in[2];
  const float* mem0  = (const float*)d_in[3];
  const float* Wxr   = (const float*)d_in[4];
  const float* Whh   = (const float*)d_in[5];
  const float* bh    = (const float*)d_in[6];
  const float* Wo    = (const float*)d_in[7];
  const float* bo    = (const float*)d_in[8];
  const float* Wp    = (const float*)d_in[9];
  const float* bp    = (const float*)d_in[10];
  float* out = (float*)d_out;

  const size_t lds_bytes = (size_t)LDS_FLOATS * sizeof(float);  // 316 KB < 320 KB/WGP
  ntm_scan_kernel<<<B_, NT, lds_bytes, stream>>>(x, ctrl0, wt0, mem0, Wxr, Whh, bh,
                                                 Wo, bo, Wp, bp, out);
}
